// CrossAttention_9302899163605
// MI455X (gfx1250) — matrix operand order repaired and checked by reading, not compiled, not run on hardware
//
#include <hip/hip_runtime.h>
#include <cstddef>
#include <cstdint>

// ---------------------------------------------------------------------------
// Problem constants (match reference)
// ---------------------------------------------------------------------------
#define CB   2
#define CNQ  2048
#define CNK  2048
#define CD   1024
#define CH   16
#define CDH  64

typedef __attribute__((ext_vector_type(8)))  float  v8f;
typedef __attribute__((ext_vector_type(8)))  __bf16 v8bf;
typedef __attribute__((ext_vector_type(16))) __bf16 v16bf;

static __device__ __forceinline__ v8f zero8() {
    v8f z = {0.f, 0.f, 0.f, 0.f, 0.f, 0.f, 0.f, 0.f};
    return z;
}

static __device__ __forceinline__ v16bf cat16(v8bf lo, v8bf hi) {
    v16bf r;
#pragma unroll
    for (int i = 0; i < 8; ++i) { r[i] = lo[i]; r[i + 8] = hi[i]; }
    return r;
}

static __device__ __forceinline__ v8f wmma_bf16(v16bf a, v16bf b, v8f c) {
    // v_wmma_f32_16x16x32_bf16 : D = A(16x32) * B(32x16) + C(16x16 f32)
    return __builtin_amdgcn_wmma_f32_16x16x32_bf16(
        /*neg_a=*/false, a, /*neg_b=*/false, b,
        /*c_mod=*/(short)0, c, /*reuse_a=*/false, /*reuse_b=*/false);
}

// A-fragment (16x32 bf16, MxK), row-major source with leading dim `ld`.
// Lane l: row M=l&15, contiguous ks [16h .. 16h+15], h = l>>4.
static __device__ __forceinline__ v16bf load_a_frag(const __bf16* __restrict__ base,
                                                    int ld, int mrow, int kk, int lane) {
    const int r    = mrow + (lane & 15);
    const int half = lane >> 4;
    const __bf16* p = base + (size_t)r * ld + kk + 16 * half;
    v8bf lo = *(const v8bf*)(p);
    v8bf hi = *(const v8bf*)(p + 8);
    return cat16(lo, hi);
}

// B-fragment (32x16 bf16, KxN): B[k][n] = src[n][k] (x@W^T pattern).
// Lane l: col N = l&15, contiguous ks [16*half .. 16*half+15].
static __device__ __forceinline__ v16bf load_b_frag(const __bf16* __restrict__ base,
                                                    int ld, int ncol, int kk, int lane) {
    const int n    = ncol + (lane & 15);
    const int half = lane >> 4;
    const __bf16* p = base + (size_t)n * ld + kk + 16 * half;
    return *(const v16bf*)(p);
}

// ---------------------------------------------------------------------------
// CDNA5 async global -> LDS (ASYNCcnt path)
// ---------------------------------------------------------------------------
static __device__ __forceinline__ void async_b128(uint32_t lds_byte_off, const void* gsrc) {
    asm volatile("global_load_async_to_lds_b128 %0, %1, off"
                 :: "v"(lds_byte_off), "v"(gsrc) : "memory");
}

// Copy a 64-row x 128-byte tile (512 x 16B chunks) with 128 threads.
static __device__ __forceinline__ void async_copy_tile(uint32_t lds_base,
                                                       const __bf16* __restrict__ gsrc,
                                                       int row_stride_elems, int tid) {
#pragma unroll
    for (int i = 0; i < 4; ++i) {
        const int chunk = tid + 128 * i;
        const int row = chunk >> 3;
        const int co  = (chunk & 7) * 16;  // byte offset within 128B row
        const char* src = (const char*)(gsrc + (size_t)row * row_stride_elems) + co;
        async_b128(lds_base + (uint32_t)(row * 128 + co), src);
    }
}

// ---------------------------------------------------------------------------
// fp32 -> bf16 elementwise convert
// ---------------------------------------------------------------------------
__global__ __launch_bounds__(256) void cvt_bf16_kernel(const float* __restrict__ in,
                                                       __bf16* __restrict__ out, int n) {
    int i = blockIdx.x * blockDim.x + threadIdx.x;
    const int stride = gridDim.x * blockDim.x;
    for (; i < n; i += stride) out[i] = (__bf16)in[i];
}

// ---------------------------------------------------------------------------
// Row LayerNorm over D=1024, fp32 in -> bf16 out. One 256-thread block / row.
// ---------------------------------------------------------------------------
__global__ __launch_bounds__(256) void ln_bf16_kernel(const float* __restrict__ x,
                                                      const float* __restrict__ gamma,
                                                      const float* __restrict__ beta,
                                                      __bf16* __restrict__ out) {
    const int row = blockIdx.x;
    const int tid = threadIdx.x;
    const float4 v = ((const float4*)(x + (size_t)row * CD))[tid];

    float s  = v.x + v.y + v.z + v.w;
    float s2 = v.x * v.x + v.y * v.y + v.z * v.z + v.w * v.w;
#pragma unroll
    for (int off = 16; off > 0; off >>= 1) {
        s  += __shfl_xor(s,  off, 32);
        s2 += __shfl_xor(s2, off, 32);
    }
    __shared__ float sh[16];
    const int w = tid >> 5, lane = tid & 31;
    if (lane == 0) { sh[w] = s; sh[8 + w] = s2; }
    __syncthreads();
    float S = 0.f, S2 = 0.f;
#pragma unroll
    for (int i = 0; i < 8; ++i) { S += sh[i]; S2 += sh[8 + i]; }

    const float mu  = S * (1.0f / CD);
    const float var = S2 * (1.0f / CD) - mu * mu;
    const float r   = rsqrtf(var + 1e-6f);

    const float4 gg = ((const float4*)gamma)[tid];
    const float4 bb = ((const float4*)beta)[tid];
    __bf16* op = out + (size_t)row * CD + tid * 4;
    op[0] = (__bf16)((v.x - mu) * r * gg.x + bb.x);
    op[1] = (__bf16)((v.y - mu) * r * gg.y + bb.y);
    op[2] = (__bf16)((v.z - mu) * r * gg.z + bb.z);
    op[3] = (__bf16)((v.w - mu) * r * gg.w + bb.w);
}

// ---------------------------------------------------------------------------
// GEMM: C[M,N] = A_bf16[M,K] @ W_bf16[N,K]^T  (+ epilogue)
// 256 threads = 8 waves, block tile 128x128, wave tile 32x64.
// MODE 0: out_f32 = acc + bias[col]
// MODE 1: KV split: col<CD -> k_raw f32; col>=CD -> vT_bf16 [B][H][DH][NK]
// ---------------------------------------------------------------------------
template <int MODE>
__global__ __launch_bounds__(256) void gemm_bf16_kernel(
    const __bf16* __restrict__ A, const __bf16* __restrict__ W,
    float* __restrict__ outf, const float* __restrict__ bias,
    float* __restrict__ kraw, __bf16* __restrict__ vT,
    int M, int N, int K) {

    const int lane = threadIdx.x & 31;
    const int w    = threadIdx.x >> 5;
    const int wm   = w & 3;
    const int wn   = w >> 2;
    const int mbase = blockIdx.x * 128 + wm * 32;
    const int nbase = blockIdx.y * 128 + wn * 64;

    v8f acc[2][4];
#pragma unroll
    for (int i = 0; i < 2; ++i)
#pragma unroll
        for (int j = 0; j < 4; ++j) acc[i][j] = zero8();

#pragma unroll 2
    for (int kk = 0; kk < K; kk += 32) {
        v16bf a0 = load_a_frag(A, K, mbase,      kk, lane);
        v16bf a1 = load_a_frag(A, K, mbase + 16, kk, lane);
        v16bf b0 = load_b_frag(W, K, nbase,      kk, lane);
        v16bf b1 = load_b_frag(W, K, nbase + 16, kk, lane);
        v16bf b2 = load_b_frag(W, K, nbase + 32, kk, lane);
        v16bf b3 = load_b_frag(W, K, nbase + 48, kk, lane);
        acc[0][0] = wmma_bf16(a0, b0, acc[0][0]);
        acc[0][1] = wmma_bf16(a0, b1, acc[0][1]);
        acc[0][2] = wmma_bf16(a0, b2, acc[0][2]);
        acc[0][3] = wmma_bf16(a0, b3, acc[0][3]);
        acc[1][0] = wmma_bf16(a1, b0, acc[1][0]);
        acc[1][1] = wmma_bf16(a1, b1, acc[1][1]);
        acc[1][2] = wmma_bf16(a1, b2, acc[1][2]);
        acc[1][3] = wmma_bf16(a1, b3, acc[1][3]);
    }

    const int half = lane >> 4, ln = lane & 15;
#pragma unroll
    for (int i = 0; i < 2; ++i) {
#pragma unroll
        for (int j = 0; j < 4; ++j) {
#pragma unroll
            for (int g = 0; g < 8; ++g) {
                const int row = mbase + 16 * i + g + 8 * half;
                const int col = nbase + 16 * j + ln;
                const float val = acc[i][j][g];
                if (MODE == 0) {
                    outf[(size_t)row * N + col] = val + bias[col];
                } else {
                    if (col < CD) {
                        kraw[(size_t)row * CD + col] = val;
                    } else {
                        const int d  = col - CD;
                        const int h  = d >> 6;
                        const int dh = d & 63;
                        const int b  = row / CNK;
                        const int nn = row - b * CNK;
                        vT[((((size_t)b * CH + h) * CDH) + dh) * CNK + nn] = (__bf16)val;
                    }
                }
            }
        }
    }
}

// ---------------------------------------------------------------------------
// Flash attention. Grid (NQ/64, H, B), 128 threads = 4 waves.
// 64-key blocks; K/V tiles staged in LDS by double-buffered async DMA
// (global_load_async_to_lds_b128, ASYNCcnt); online softmax with cross-lane
// reductions; P relayout through per-wave LDS tile.
// q, k : bf16 [B][N][D] (head h at column h*64); vT : bf16 [B][H][DH][NK].
// ---------------------------------------------------------------------------
__global__ __launch_bounds__(128) void flash_attn_kernel(
    const __bf16* __restrict__ q, const __bf16* __restrict__ k,
    const __bf16* __restrict__ vT, __bf16* __restrict__ o_out) {

    const int tid  = threadIdx.x;
    const int lane = tid & 31;
    const int w    = tid >> 5;
    const int b    = blockIdx.z;
    const int h    = blockIdx.y;
    const int mbase = blockIdx.x * 64 + w * 16;
    const int half = lane >> 4, ln = lane & 15;

    const __bf16* qp = q  + (size_t)b * CNQ * CD + h * CDH;        // ld = CD
    const __bf16* kp = k  + (size_t)b * CNK * CD + h * CDH;        // ld = CD
    const __bf16* vp = vT + (((size_t)b * CH + h) * CDH) * CNK;    // [DH][NK], ld = CNK

    // K tile: [64 keys][64 dh]; V tile: [64 dh][64 keys]; both 128B rows.
    __shared__ __attribute__((aligned(32))) __bf16 kt[2][64 * 64];
    __shared__ __attribute__((aligned(32))) __bf16 vt[2][64 * 64];
    __shared__ __attribute__((aligned(32))) __bf16 ptile_all[4][16 * 72];
    __bf16* ptile = ptile_all[w];

    const uint32_t kt_off[2] = { (uint32_t)(uintptr_t)&kt[0][0], (uint32_t)(uintptr_t)&kt[1][0] };
    const uint32_t vt_off[2] = { (uint32_t)(uintptr_t)&vt[0][0], (uint32_t)(uintptr_t)&vt[1][0] };

    v16bf qa[2];
    qa[0] = load_a_frag(qp, CD, mbase, 0,  lane);
    qa[1] = load_a_frag(qp, CD, mbase, 32, lane);

    v8f o[4];
#pragma unroll
    for (int d = 0; d < 4; ++d) o[d] = zero8();
    float m[8], l[8];
#pragma unroll
    for (int g = 0; g < 8; ++g) { m[g] = -__builtin_inff(); l[g] = 0.f; }

    const float scale = 0.125f;  // DH^-0.5

    // Preload first K/V tiles into buffer 0 (8 async b128 per thread).
    async_copy_tile(kt_off[0], kp, CD, tid);
    async_copy_tile(vt_off[0], vp, CNK, tid);

#pragma unroll 1
    for (int j0 = 0, it = 0; j0 < CNK; j0 += 64, ++it) {
        const int cur = it & 1;
        const bool pre = (j0 + 64) < CNK;
        if (pre) {
            // DMA next tile while we compute this one.
            async_copy_tile(kt_off[cur ^ 1], kp + (size_t)(j0 + 64) * CD, CD, tid);
            async_copy_tile(vt_off[cur ^ 1], vp + (j0 + 64), CNK, tid);
            asm volatile("s_wait_asynccnt 0x8" ::: "memory");  // older batch done
        } else {
            asm volatile("s_wait_asynccnt 0x0" ::: "memory");
        }
        __syncthreads();  // all waves' async writes for `cur` visible

        const __bf16* ktc = &kt[cur][0];
        const __bf16* vtc = &vt[cur][0];

        // S = Q @ K^T : 16 rows x 64 keys (4 accumulators, 8 WMMA)
        v8f st[4];
#pragma unroll
        for (int t = 0; t < 4; ++t) st[t] = zero8();
#pragma unroll
        for (int s = 0; s < 2; ++s) {
#pragma unroll
            for (int t = 0; t < 4; ++t) {
                const v16bf kb = *(const v16bf*)(ktc + (16 * t + ln) * 64 + 32 * s + 16 * half);
                st[t] = wmma_bf16(qa[s], kb, st[t]);
            }
        }

        // Online softmax over 64 keys
        float p[4][8];
#pragma unroll
        for (int g = 0; g < 8; ++g) {
            float mt = fmaxf(fmaxf(st[0][g], st[1][g]), fmaxf(st[2][g], st[3][g])) * scale;
#pragma unroll
            for (int off = 1; off < 16; off <<= 1) mt = fmaxf(mt, __shfl_xor(mt, off, 32));
            const float mnew  = fmaxf(m[g], mt);
            const float alpha = __expf(m[g] - mnew);
            float rs = 0.f;
#pragma unroll
            for (int t = 0; t < 4; ++t) {
                p[t][g] = __expf(st[t][g] * scale - mnew);
                rs += p[t][g];
            }
#pragma unroll
            for (int off = 1; off < 16; off <<= 1) rs += __shfl_xor(rs, off, 32);
            l[g] = l[g] * alpha + rs;
            m[g] = mnew;
#pragma unroll
            for (int d = 0; d < 4; ++d) o[d][g] = o[d][g] * alpha;
        }

        // P (C/D layout) -> per-wave LDS [16 rows][64 keys], stride 72 elems
        __syncthreads();
#pragma unroll
        for (int g = 0; g < 8; ++g) {
            const int r = g + 8 * half;
#pragma unroll
            for (int t = 0; t < 4; ++t)
                ptile[r * 72 + 16 * t + ln] = (__bf16)p[t][g];
        }
        __syncthreads();

        // O += P @ V : two A-fragments of P, V^T rows contiguous over keys
#pragma unroll
        for (int s = 0; s < 2; ++s) {
            v16bf pa;
            {
                const __bf16* pp = ptile + ln * 72 + 32 * s + 16 * half;
                v8bf lo = *(const v8bf*)(pp);
                v8bf hi = *(const v8bf*)(pp + 8);
                pa = cat16(lo, hi);
            }
#pragma unroll
            for (int d = 0; d < 4; ++d) {
                const v16bf vb = *(const v16bf*)(vtc + (16 * d + ln) * 64 + 32 * s + 16 * half);
                o[d] = wmma_bf16(pa, vb, o[d]);
            }
        }
        __syncthreads();  // all reads of `cur` done before it is overwritten
    }

    // Normalize and store bf16 [B][NQ][D]
    __bf16* op = o_out + (size_t)b * CNQ * CD + h * CDH;
#pragma unroll
    for (int d = 0; d < 4; ++d)
#pragma unroll
        for (int g = 0; g < 8; ++g) {
            const int row = mbase + g + 8 * half;
            op[(size_t)row * CD + 16 * d + ln] = (__bf16)(o[d][g] / l[g]);
        }
}

// ---------------------------------------------------------------------------
// Launcher
// ---------------------------------------------------------------------------
extern "C" void kernel_launch(void* const* d_in, const int* in_sizes, int n_in,
                              void* d_out, int out_size, void* d_ws, size_t ws_size,
                              hipStream_t stream) {
    const float* x    = (const float*)d_in[0];
    const float* ctx  = (const float*)d_in[1];
    const float* gq   = (const float*)d_in[2];
    const float* bq   = (const float*)d_in[3];
    const float* gk   = (const float*)d_in[4];
    const float* bk   = (const float*)d_in[5];
    const float* Wkv  = (const float*)d_in[6];
    const float* Wout = (const float*)d_in[7];
    const float* bout = (const float*)d_in[8];

    char* ws = (char*)d_ws;
    auto carve = [&](size_t bytes) -> char* {
        char* p = ws;
        ws += (bytes + 255) & ~(size_t)255;
        return p;
    };

    const size_t NTOK = (size_t)CB * CNQ * CD;   // 4M elements
    __bf16* q_b    = (__bf16*)carve(NTOK * 2);
    __bf16* ctx_b  = (__bf16*)carve(NTOK * 2);
    __bf16* k_b    = (__bf16*)carve(NTOK * 2);
    __bf16* vT_b   = (__bf16*)carve(NTOK * 2);   // [B][H][DH][NK]
    __bf16* attn_b = (__bf16*)carve(NTOK * 2);
    __bf16* wkv_b  = (__bf16*)carve((size_t)2 * CD * CD * 2);
    __bf16* wout_b = (__bf16*)carve((size_t)CD * CD * 2);
    float*  k_raw  = (float*)carve(NTOK * 4);

    // 1) fp32 -> bf16 converts
    cvt_bf16_kernel<<<2048, 256, 0, stream>>>(ctx,  ctx_b,  (int)NTOK);
    cvt_bf16_kernel<<<2048, 256, 0, stream>>>(Wkv,  wkv_b,  2 * CD * CD);
    cvt_bf16_kernel<<<1024, 256, 0, stream>>>(Wout, wout_b, CD * CD);

    // 2) q = LN(x)
    ln_bf16_kernel<<<CB * CNQ, 256, 0, stream>>>(x, gq, bq, q_b);

    // 3) KV projection: [4096,1024] @ [2048,1024]^T -> k_raw f32 / vT bf16
    gemm_bf16_kernel<1><<<dim3((CB * CNK) / 128, (2 * CD) / 128), 256, 0, stream>>>(
        ctx_b, wkv_b, nullptr, nullptr, k_raw, vT_b, CB * CNK, 2 * CD, CD);

    // 4) k = LN(k_raw)
    ln_bf16_kernel<<<CB * CNK, 256, 0, stream>>>(k_raw, gk, bk, k_b);

    // 5) flash attention -> attn_b bf16 [B][NQ][D]
    flash_attn_kernel<<<dim3(CNQ / 64, CH, CB), 128, 0, stream>>>(q_b, k_b, vT_b, attn_b);

    // 6) output projection + bias -> d_out f32
    gemm_bf16_kernel<0><<<dim3((CB * CNQ) / 128, CD / 128), 256, 0, stream>>>(
        attn_b, wout_b, (float*)d_out, bout, nullptr, nullptr, CB * CNQ, CD, CD);
}
